// LSTMLayer_80650895884766
// MI455X (gfx1250) — matrix operand order repaired and checked by reading, not compiled, not run on hardware
//
#include <hip/hip_runtime.h>
#include <stdint.h>

// LSTM on gfx1250: bf16 WMMA (v_wmma_f32_16x16x32_bf16) for all GEMMs.
// Recurrence runs in ONE persistent kernel (128 resident blocks) with a
// device-scope generation barrier between timesteps; cell state c is
// register-resident per block. INPUT=512, HIDDEN=1024, OUTPUT=512, B=64, T=512.
#define INPUT_   512
#define HIDDEN_  1024
#define OUTPUT_  512
#define BATCH_   64
#define TSTEPS_  512
#define NBLK_    128    // persistent blocks (trivially co-resident)

typedef __attribute__((ext_vector_type(16))) __bf16        v16bf;
typedef __attribute__((ext_vector_type(8)))  float         v8f;
typedef __attribute__((ext_vector_type(4)))  unsigned int  u32x4;

union Frag16 { v16bf v; u32x4 q[2]; };

__device__ __forceinline__ unsigned short f32_bf16(float f) {
  unsigned int u = __float_as_uint(f);
  unsigned int r = u + 0x7FFFu + ((u >> 16) & 1u);   // round to nearest even
  return (unsigned short)(r >> 16);
}

__device__ __forceinline__ float sigmoidf_(float x) {
  return 1.0f / (1.0f + __expf(-x));
}

// --------------------------------------------------------------- grid barrier
// Generation-counter barrier at agent scope. Release on arrive makes this
// block's h_{t+1} stores device-visible; acquire on exit invalidates the WGP
// caches before reading other blocks' h.
__device__ __forceinline__ void grid_barrier(unsigned* cnt, unsigned* gen) {
  __syncthreads();
  if (threadIdx.x == 0) {
    unsigned g = __hip_atomic_load(gen, __ATOMIC_RELAXED, __HIP_MEMORY_SCOPE_AGENT);
    unsigned arrived = __hip_atomic_fetch_add(cnt, 1u, __ATOMIC_ACQ_REL,
                                              __HIP_MEMORY_SCOPE_AGENT);
    if (arrived == NBLK_ - 1u) {
      __hip_atomic_store(cnt, 0u, __ATOMIC_RELAXED, __HIP_MEMORY_SCOPE_AGENT);
      __hip_atomic_store(gen, g + 1u, __ATOMIC_RELEASE, __HIP_MEMORY_SCOPE_AGENT);
    } else {
      while (__hip_atomic_load(gen, __ATOMIC_ACQUIRE, __HIP_MEMORY_SCOPE_AGENT) == g)
        __builtin_amdgcn_s_sleep(2);
    }
  }
  __syncthreads();
}

// ---------------------------------------------------------------- init state
__global__ void init_state_kernel(unsigned short* __restrict__ h0,
                                  unsigned* __restrict__ bar) {
  int i = blockIdx.x * blockDim.x + threadIdx.x;   // 65536 = B*H
  h0[i] = 0;
  if (i < 2) bar[i] = 0;
}

// ---------------------------------------------------------------- f32 -> bf16
__global__ void cvt_bf16_kernel(const float* __restrict__ src,
                                unsigned short* __restrict__ dst, size_t n) {
  size_t i = (size_t)blockIdx.x * blockDim.x + threadIdx.x;
  if (i < n) dst[i] = f32_bf16(src[i]);
}

// ----------------------------------------------- transpose + convert to bf16
// src: [R][C] f32   dst: [C][R] bf16   (dst[c*R + r] = src[r*C + c])
__global__ void transpose_bf16_kernel(const float* __restrict__ src,
                                      unsigned short* __restrict__ dst,
                                      int R, int C) {
  size_t idx = (size_t)blockIdx.x * blockDim.x + threadIdx.x;
  size_t total = (size_t)R * (size_t)C;
  if (idx >= total) return;
  size_t c = idx / (size_t)R;
  size_t r = idx % (size_t)R;
  dst[idx] = f32_bf16(src[r * (size_t)C + c]);
}

// ------------------------------------------------------- persistent recurrence
// Block = 128 threads = 4 waves; wave g owns gate g (i,f,o,c). Block bid owns
// 16x16 tiles {bid, bid+128} of the [64 x 1024] gate matrix for ALL timesteps,
// so its c-tile stays in registers. Per step, per tile:
//   acc = bias + h_t @ W_h[g] + x[:,t,:] @ W_x[g]     (48 WMMAs per gate-tile)
// then gates meet in LDS and the cell update writes h_{t+1} (bf16) to the ring.
//
// A-fragment (16x32 bf16, ISA 7.12.2): lane l: m=l&15, kb=8*(l>>4);
//   elems 0..7  -> K = kk+kb..+7 (b128);  elems 8..15 -> K = kk+16+kb..+7 (b128)
// B-fragment (32x16 bf16): lane l: n=l&15, same K order as A:
//   elems 0..7 -> K = kk+kb..+7, elems 8..15 -> K = kk+16+kb..+7 (two b128, N-major).
__global__ __launch_bounds__(128)
void lstm_persistent_kernel(unsigned short* __restrict__ hs,      // (T+1)*B*H bf16
                            const unsigned short* __restrict__ xbf, // [B][T][I] bf16
                            const unsigned short* __restrict__ WTh, // 4 x [H][H] n-major
                            const unsigned short* __restrict__ WTx, // 4 x [H][I] n-major
                            const float* __restrict__ b_i, const float* __restrict__ b_f,
                            const float* __restrict__ b_o, const float* __restrict__ b_c,
                            unsigned* __restrict__ bar_cnt,
                            unsigned* __restrict__ bar_gen) {
  __shared__ float lds[4 * 256];

  const int tid   = threadIdx.x;
  const int lane  = tid & 31;
  const int g     = tid >> 5;                 // gate 0..3 (i,f,o,c)
  const int lan16 = lane & 15;
  const int hi    = lane >> 4;
  const int kbA   = 8 * hi;
  const int kbB   = 16 * hi;

  const float* bias = (g == 0) ? b_i : (g == 1) ? b_f : (g == 2) ? b_o : b_c;

  // loop-invariant per-tile state
  int m0s[2], n0s[2];
  const unsigned short* Whg_row[2];
  const unsigned short* Wxg_row[2];
  float bg[2];
  size_t hA_off[2];      // + t*B*H
  size_t xA_off[2];      // + t*I
#pragma unroll
  for (int j = 0; j < 2; ++j) {
    int tile = (int)blockIdx.x + j * NBLK_;   // 0..255
    int mt = tile >> 6, nt = tile & 63;
    m0s[j] = mt * 16;
    n0s[j] = nt * 16;
    int m  = m0s[j] + lan16;
    int nB = n0s[j] + lan16;
    Whg_row[j] = WTh + (size_t)g * HIDDEN_ * HIDDEN_ + (size_t)nB * HIDDEN_;
    Wxg_row[j] = WTx + (size_t)g * HIDDEN_ * INPUT_  + (size_t)nB * INPUT_;
    bg[j]      = bias[n0s[j] + lan16];
    hA_off[j]  = (size_t)m * HIDDEN_;
    xA_off[j]  = (size_t)m * (TSTEPS_ * INPUT_);
    // warm L2/WGP$ for this block's permanent weight rows
    __builtin_prefetch(Whg_row[j], 0, 1);
    __builtin_prefetch(Wxg_row[j], 0, 1);
  }

  float creg[2][2] = {{0.0f, 0.0f}, {0.0f, 0.0f}};   // register-resident cell state

  for (int t = 0; t < TSTEPS_; ++t) {
    v8f acc[2];
#pragma unroll
    for (int j = 0; j < 2; ++j) {
      v8f a_;
#pragma unroll
      for (int r = 0; r < 8; ++r) a_[r] = bg[j];

      // recurrent part: h_t @ W_h[g], K = HIDDEN
      const unsigned short* hA = hs + (size_t)t * (BATCH_ * HIDDEN_) + hA_off[j];
      const unsigned short* Wg = Whg_row[j];
#pragma unroll 4
      for (int kk = 0; kk < HIDDEN_; kk += 32) {
        Frag16 a, b;
        a.q[0] = *(const u32x4*)(hA + kk + kbA);
        a.q[1] = *(const u32x4*)(hA + kk + 16 + kbA);
        b.q[0] = *(const u32x4*)(Wg + kk + kbA);
        b.q[1] = *(const u32x4*)(Wg + kk + 16 + kbA);
        a_ = __builtin_amdgcn_wmma_f32_16x16x32_bf16(false, a.v, false, b.v,
                                                     (short)0, a_, false, false);
      }
      // input part: x[:, t, :] @ W_x[g], K = INPUT (row stride T*I)
      const unsigned short* xA = xbf + xA_off[j] + (size_t)t * INPUT_;
      const unsigned short* Wx = Wxg_row[j];
#pragma unroll 4
      for (int kk = 0; kk < INPUT_; kk += 32) {
        Frag16 a, b;
        a.q[0] = *(const u32x4*)(xA + kk + kbA);
        a.q[1] = *(const u32x4*)(xA + kk + 16 + kbA);
        b.q[0] = *(const u32x4*)(Wx + kk + kbA);
        b.q[1] = *(const u32x4*)(Wx + kk + 16 + kbA);
        a_ = __builtin_amdgcn_wmma_f32_16x16x32_bf16(false, a.v, false, b.v,
                                                     (short)0, a_, false, false);
      }
      acc[j] = a_;
    }

    unsigned short* hOut = hs + (size_t)(t + 1) * (BATCH_ * HIDDEN_);
#pragma unroll
    for (int j = 0; j < 2; ++j) {
      // exchange the 4 gate tiles through LDS
#pragma unroll
      for (int r = 0; r < 8; ++r) lds[g * 256 + r * 32 + lane] = acc[j][r];
      __syncthreads();
      // elementwise cell update: 256 tile elements over 128 threads
#pragma unroll
      for (int pp = 0; pp < 2; ++pp) {
        int p = tid + pp * 128;
        int l = p & 31;
        int r = p >> 5;
        float gi = lds[0 * 256 + r * 32 + l];
        float gf = lds[1 * 256 + r * 32 + l];
        float go = lds[2 * 256 + r * 32 + l];
        float gc = lds[3 * 256 + r * 32 + l];
        float it = sigmoidf_(gi);
        float ft = sigmoidf_(gf);
        float ot = sigmoidf_(go);
        float cd = tanhf(gc);
        float cn = ft * creg[j][pp] + it * cd;      // c stays in registers
        creg[j][pp] = cn;
        int mrow = m0s[j] + r + 8 * (l >> 4);       // C-layout: M = r + 8*(lane>>4)
        int ncol = n0s[j] + (l & 15);
        hOut[mrow * HIDDEN_ + ncol] = f32_bf16(ot * tanhf(cn));
      }
      __syncthreads();                              // protect LDS reuse for tile 1
    }

    grid_barrier(bar_cnt, bar_gen);                 // h_{t+1} visible device-wide
  }
}

// ------------------------------------------------------------ output proj
// ys[t*64+b] = hs[t+1] row @ W_hy + b_y ; store into out[b][t][o].
// One wave per 16x16 tile: 2048 M-tiles x 32 N-tiles = 65536 waves.
__global__ __launch_bounds__(256)
void out_proj_kernel(const unsigned short* __restrict__ hs,   // (T+1)*B*H bf16
                     const unsigned short* __restrict__ WThy, // [O][H] n-major
                     const float* __restrict__ by,
                     float* __restrict__ out) {
  const int w    = blockIdx.x * 8 + (threadIdx.x >> 5);
  const int lane = threadIdx.x & 31;
  const int mt   = w >> 5;        // 0..2047
  const int nt   = w & 31;        // 0..31
  const int m0   = mt * 16, n0 = nt * 16;

  const int lan16 = lane & 15;
  const int hi    = lane >> 4;
  const int kbA   = 8 * hi;
  const int kbB   = 16 * hi;
  const int m     = m0 + lan16;

  v8f acc;
  {
    float bgv = by[n0 + lan16];
#pragma unroll
    for (int r = 0; r < 8; ++r) acc[r] = bgv;
  }

  // row (t*64+b) lives at plane t+1 -> linear offset (m + 64) rows of H
  const unsigned short* hA = hs + (size_t)(m + BATCH_) * HIDDEN_;
  const unsigned short* Wg = WThy + (size_t)(n0 + lan16) * HIDDEN_;
#pragma unroll 4
  for (int kk = 0; kk < HIDDEN_; kk += 32) {
    Frag16 a, b;
    a.q[0] = *(const u32x4*)(hA + kk + kbA);
    a.q[1] = *(const u32x4*)(hA + kk + 16 + kbA);
    b.q[0] = *(const u32x4*)(Wg + kk + kbA);
    b.q[1] = *(const u32x4*)(Wg + kk + 16 + kbA);
    acc = __builtin_amdgcn_wmma_f32_16x16x32_bf16(false, a.v, false, b.v,
                                                  (short)0, acc, false, false);
  }

#pragma unroll
  for (int r = 0; r < 8; ++r) {
    int mrow = m0 + r + 8 * hi;               // = t*64 + b
    int bb = mrow & 63;
    int tt = mrow >> 6;
    out[(size_t)bb * (TSTEPS_ * OUTPUT_) + (size_t)tt * OUTPUT_ + n0 + lan16] = acc[r];
  }
}

// ---------------------------------------------------------------------------
extern "C" void kernel_launch(void* const* d_in, const int* in_sizes, int n_in,
                              void* d_out, int out_size, void* d_ws, size_t ws_size,
                              hipStream_t stream) {
  (void)in_sizes; (void)n_in; (void)out_size; (void)ws_size;
  const float* x  = (const float*)d_in[0];
  const float* Wx[4]   = { (const float*)d_in[1], (const float*)d_in[4],
                           (const float*)d_in[7], (const float*)d_in[10] };
  const float* Wh[4]   = { (const float*)d_in[2], (const float*)d_in[5],
                           (const float*)d_in[8], (const float*)d_in[11] };
  const float* bias[4] = { (const float*)d_in[3], (const float*)d_in[6],
                           (const float*)d_in[9], (const float*)d_in[12] };
  const float* Why = (const float*)d_in[13];
  const float* by  = (const float*)d_in[14];
  float* out = (float*)d_out;

  char* ws = (char*)d_ws;
  size_t off = 0;
  auto alloc = [&](size_t bytes) -> void* {
    void* p = ws + off;
    off = (off + bytes + 255) & ~(size_t)255;
    return p;
  };
  unsigned short* xbf  = (unsigned short*)alloc((size_t)BATCH_ * TSTEPS_ * INPUT_ * 2);  // 33.6 MB
  unsigned short* WTx  = (unsigned short*)alloc((size_t)4 * HIDDEN_ * INPUT_ * 2);       //  4.2 MB
  unsigned short* WTh  = (unsigned short*)alloc((size_t)4 * HIDDEN_ * HIDDEN_ * 2);      //  8.4 MB
  unsigned short* WThy = (unsigned short*)alloc((size_t)OUTPUT_ * HIDDEN_ * 2);          //  1.0 MB
  unsigned short* hs   = (unsigned short*)alloc((size_t)(TSTEPS_ + 1) * BATCH_ * HIDDEN_ * 2); // 67 MB
  unsigned*       bar  = (unsigned*)alloc(2 * sizeof(unsigned));

  // h0 = 0, barrier counters = 0
  init_state_kernel<<<dim3((BATCH_ * HIDDEN_) / 256), dim3(256), 0, stream>>>(hs, bar);

  // x -> bf16
  {
    size_t n = (size_t)BATCH_ * TSTEPS_ * INPUT_;
    cvt_bf16_kernel<<<dim3((unsigned)((n + 255) / 256)), dim3(256), 0, stream>>>(x, xbf, n);
  }
  // weights -> bf16, transposed to N-major (contiguous-K B fragments)
  for (int g = 0; g < 4; ++g) {
    transpose_bf16_kernel<<<dim3((INPUT_ * HIDDEN_ + 255) / 256), dim3(256), 0, stream>>>(
        Wx[g], WTx + (size_t)g * HIDDEN_ * INPUT_, INPUT_, HIDDEN_);
    transpose_bf16_kernel<<<dim3((HIDDEN_ * HIDDEN_ + 255) / 256), dim3(256), 0, stream>>>(
        Wh[g], WTh + (size_t)g * HIDDEN_ * HIDDEN_, HIDDEN_, HIDDEN_);
  }
  transpose_bf16_kernel<<<dim3((HIDDEN_ * OUTPUT_ + 255) / 256), dim3(256), 0, stream>>>(
      Why, WThy, HIDDEN_, OUTPUT_);

  // whole recurrence in one persistent kernel (device barrier between steps)
  lstm_persistent_kernel<<<dim3(NBLK_), dim3(128), 0, stream>>>(
      hs, xbf, WTh, WTx, bias[0], bias[1], bias[2], bias[3], bar, bar + 1);

  // output projection over all timesteps
  out_proj_kernel<<<dim3(8192), dim3(256), 0, stream>>>(hs, WThy, by, out);
}
